// Conv3dBlock_10307921510639
// MI455X (gfx1250) — matrix operand order repaired and checked by reading, not compiled, not run on hardware
//
#include <hip/hip_runtime.h>
#include <hip/hip_bf16.h>
#include <math.h>

// Problem constants (match reference)
#define Bn 2
#define Cn 256
#define Tn 128
#define Mn 128
#define Hn 32
#define Wn 32
#define NROWS (Bn*Tn*Cn)          // 65536 active rows
#define NPOS  (Bn*Tn*Hn*Wn)       // 262144 positions per channel (BN count)
#define NBLK_CONV (Bn*Tn*Hn)      // 8192

typedef _Float16 v16h __attribute__((ext_vector_type(16)));
typedef float    v8f  __attribute__((ext_vector_type(8)));
typedef float    f4   __attribute__((ext_vector_type(4)));

// ---------------- prep: pw_w -> f16, inv map init ----------------
__global__ void prep_kernel(const float* __restrict__ pw_w,
                            _Float16* __restrict__ pw_wh,
                            int* __restrict__ inv) {
  int tid = blockIdx.x * blockDim.x + threadIdx.x;
  if (tid < Mn * Mn) pw_wh[tid] = (_Float16)pw_w[tid];
  if (tid < Hn * Wn) inv[tid] = -1;
}

__global__ void scatter_kernel(const int* __restrict__ row_idx,
                               const int* __restrict__ col_idx,
                               int* __restrict__ inv) {
  int c = threadIdx.x;  // 256 threads
  inv[row_idx[c] * Wn + col_idx[c]] = c;
}

// ---------------- depthwise conv + bias + stat partials + active gather ----------------
// block = (b*T + t)*H + h ; 256 threads: m = tid&127, w-half = tid>>7 (16 w each)
__global__ __launch_bounds__(256) void conv_kernel(
    const float* __restrict__ x, const float* __restrict__ conv_w,
    const float* __restrict__ conv_b, const int* __restrict__ inv,
    float* __restrict__ yact, float* __restrict__ ps1, float* __restrict__ ps2) {
  __shared__ float ldsw[27 * Mn];
  __shared__ float red[4 * Mn];
  const int tid = threadIdx.x;
  for (int i = tid; i < 27 * Mn; i += 256) {
    int m = i / 27, k = i % 27;
    ldsw[k * Mn + m] = conv_w[i];            // transpose: conflict-free per-m reads
  }
  __syncthreads();

  const int blk = blockIdx.x;
  const int h = blk & (Hn - 1);
  const int t = (blk >> 5) & (Tn - 1);
  const int b = blk >> 12;
  const int m = tid & (Mn - 1);
  const int wh = tid >> 7;
  const int w0 = wh * 16;

  float acc[16];
  const float bias = conv_b[m];
#pragma unroll
  for (int w = 0; w < 16; ++w) acc[w] = bias;

#pragma unroll
  for (int dh = 0; dh < 3; ++dh) {
    const int hp = h + dh - 1;
    int cv[18];
#pragma unroll
    for (int j = 0; j < 18; ++j) {
      int wp = w0 + j - 1;
      cv[j] = (hp >= 0 && hp < Hn && wp >= 0 && wp < Wn) ? inv[hp * Wn + wp] : -1;
    }
#pragma unroll
    for (int dt = 0; dt < 3; ++dt) {
      const int tp = t + dt - 2;            // causal pad (KT-1, 0)
      if (tp < 0) continue;
      float val[18];
#pragma unroll
      for (int j = 0; j < 18; ++j)
        val[j] = (cv[j] >= 0) ? x[((b * Cn + cv[j]) * Tn + tp) * Mn + m] : 0.f;
      const int kb = dt * 9 + dh * 3;
      const float k0 = ldsw[(kb + 0) * Mn + m];
      const float k1 = ldsw[(kb + 1) * Mn + m];
      const float k2 = ldsw[(kb + 2) * Mn + m];
#pragma unroll
      for (int w = 0; w < 16; ++w)
        acc[w] = fmaf(val[w], k0, fmaf(val[w + 1], k1, fmaf(val[w + 2], k2, acc[w])));
    }
  }

  // per-channel partial sums (deterministic: fixed block layout)
  float s1 = 0.f, s2 = 0.f;
#pragma unroll
  for (int w = 0; w < 16; ++w) { s1 += acc[w]; s2 += acc[w] * acc[w]; }
  red[wh * Mn + m] = s1;
  red[2 * Mn + wh * Mn + m] = s2;
  __syncthreads();
  if (tid < Mn) {
    ps1[blk * Mn + tid] = red[tid] + red[Mn + tid];
    ps2[blk * Mn + tid] = red[2 * Mn + tid] + red[3 * Mn + tid];
  }

  // store conv output only at active positions, row-major [row][m]
#pragma unroll
  for (int w = 0; w < 16; ++w) {
    const int c = inv[h * Wn + w0 + w];
    if (c >= 0) yact[((b * Tn + t) * Cn + c) * Mn + m] = acc[w];
  }
}

// ---------------- finalize stats: fold into per-channel scale/shift ----------------
__global__ __launch_bounds__(256) void stats_kernel(
    const float* __restrict__ ps1, const float* __restrict__ ps2,
    const float* __restrict__ gamma, const float* __restrict__ beta,
    float* __restrict__ stats) {
  __shared__ float r1[256], r2[256];
  const int m = blockIdx.x, tid = threadIdx.x;
  float s1 = 0.f, s2 = 0.f;
  for (int i = tid; i < NBLK_CONV; i += 256) { s1 += ps1[i * Mn + m]; s2 += ps2[i * Mn + m]; }
  r1[tid] = s1; r2[tid] = s2;
  __syncthreads();
  for (int s = 128; s > 0; s >>= 1) {
    if (tid < s) { r1[tid] += r1[tid + s]; r2[tid] += r2[tid + s]; }
    __syncthreads();
  }
  if (tid == 0) {
    const float invN = 1.f / (float)NPOS;
    const float mean = r1[0] * invN;
    const float var  = r2[0] * invN - mean * mean;
    const float sc   = rsqrtf(var + 1e-5f) * gamma[m];
    stats[m]      = sc;                   // scale
    stats[Mn + m] = beta[m] - mean * sc;  // shift
  }
}

// ---------------- fused norm+GELU + WMMA pointwise GEMM + residual epilogue ----------------
// 128 threads = 4 waves; each wave owns a 16-row tile, full N=128 (8 n-tiles), K=128 (4 steps)
__global__ __launch_bounds__(128) void gemm_kernel(
    const float* __restrict__ yact, const _Float16* __restrict__ pw_wh,
    const float* __restrict__ stats, const float* __restrict__ pw_b,
    const float* __restrict__ x, float* __restrict__ out) {
  const int lane = threadIdx.x & 31;
  const int wid  = threadIdx.x >> 5;
  const int row0 = (blockIdx.x * 4 + wid) * 16;
  const int lrow = lane & 15;
  const int hi   = lane >> 4;  // 0: lanes 0-15, 1: lanes 16-31

  v8f acc[8] = {};
  const float* yrow = yact + (size_t)(row0 + lrow) * Mn;

#pragma unroll
  for (int kk = 0; kk < 4; ++kk) {
    // A layout (16-bit A 16x32): lane group selects K base, 16 contiguous K per lane
    const int mb = kk * 32 + hi * 16;
    f4 y0 = *(const f4*)(yrow + mb);
    f4 y1 = *(const f4*)(yrow + mb + 4);
    f4 y2 = *(const f4*)(yrow + mb + 8);
    f4 y3 = *(const f4*)(yrow + mb + 12);
    f4 sc0 = *(const f4*)(stats + mb);
    f4 sc1 = *(const f4*)(stats + mb + 4);
    f4 sc2 = *(const f4*)(stats + mb + 8);
    f4 sc3 = *(const f4*)(stats + mb + 12);
    f4 sh0 = *(const f4*)(stats + Mn + mb);
    f4 sh1 = *(const f4*)(stats + Mn + mb + 4);
    f4 sh2 = *(const f4*)(stats + Mn + mb + 8);
    f4 sh3 = *(const f4*)(stats + Mn + mb + 12);

    v16h a;
#pragma unroll
    for (int j = 0; j < 4; ++j) {
      float v0 = fmaf(y0[j], sc0[j], sh0[j]);
      a[j]      = (_Float16)(0.5f * v0 * (1.f + erff(v0 * 0.70710678118f)));
      float v1 = fmaf(y1[j], sc1[j], sh1[j]);
      a[4 + j]  = (_Float16)(0.5f * v1 * (1.f + erff(v1 * 0.70710678118f)));
      float v2 = fmaf(y2[j], sc2[j], sh2[j]);
      a[8 + j]  = (_Float16)(0.5f * v2 * (1.f + erff(v2 * 0.70710678118f)));
      float v3 = fmaf(y3[j], sc3[j], sh3[j]);
      a[12 + j] = (_Float16)(0.5f * v3 * (1.f + erff(v3 * 0.70710678118f)));
    }

    // B layout (16-bit B 32x16): lane = N column, 16 contiguous K halves (32B) per lane
    const int kb = kk * 32 + hi * 16;
#pragma unroll
    for (int nt = 0; nt < 8; ++nt) {
      v16h bmat = *(const v16h*)(pw_wh + (nt * 16 + lrow) * Mn + kb);
      acc[nt] = __builtin_amdgcn_wmma_f32_16x16x32_f16(
          false, a, false, bmat, (short)0, acc[nt], false, false);
    }
  }

  // Epilogue: D(lane, vgpr i) -> row = row0 + i + hi*8, col n = nt*16 + lrow
#pragma unroll
  for (int nt = 0; nt < 8; ++nt) {
#pragma unroll
    for (int i = 0; i < 8; ++i) {
      const int r = row0 + i + hi * 8;           // r = (b*T + t)*C + c
      const int n = nt * 16 + lrow;
      const int b = r >> 15;
      const int t = (r >> 8) & (Tn - 1);
      const int c = r & (Cn - 1);
      const int idx = (((b * Cn + c) * Tn) + t) * Mn + n;  // (B,C,T,M) layout
      out[idx] = acc[nt][i] + pw_b[n] + x[idx];  // residual grid gather == x
    }
  }
}

extern "C" void kernel_launch(void* const* d_in, const int* in_sizes, int n_in,
                              void* d_out, int out_size, void* d_ws, size_t ws_size,
                              hipStream_t stream) {
  const float* x       = (const float*)d_in[0];
  const int*   row_idx = (const int*)d_in[1];
  const int*   col_idx = (const int*)d_in[2];
  const float* conv_w  = (const float*)d_in[3];
  const float* conv_b  = (const float*)d_in[4];
  const float* pw_w    = (const float*)d_in[5];
  const float* pw_b    = (const float*)d_in[6];
  const float* gamma   = (const float*)d_in[7];
  const float* beta    = (const float*)d_in[8];
  float* out = (float*)d_out;

  // workspace layout (all offsets 32B-aligned)
  float*    yact  = (float*)d_ws;                    // 65536*128 f32 = 32 MiB
  float*    ps1   = yact + (size_t)NROWS * Mn;       // 8192*128 f32  = 4 MiB
  float*    ps2   = ps1 + (size_t)NBLK_CONV * Mn;    // 4 MiB
  float*    stats = ps2 + (size_t)NBLK_CONV * Mn;    // 256 f32 (scale, shift)
  int*      inv   = (int*)(stats + 2 * Mn);          // 1024 i32
  _Float16* pw_wh = (_Float16*)(inv + Hn * Wn);      // 16384 f16 = 32 KiB

  prep_kernel<<<(Mn * Mn + 255) / 256, 256, 0, stream>>>(pw_w, pw_wh, inv);
  scatter_kernel<<<1, Cn, 0, stream>>>(row_idx, col_idx, inv);
  conv_kernel<<<NBLK_CONV, 256, 0, stream>>>(x, conv_w, conv_b, inv, yact, ps1, ps2);
  stats_kernel<<<Mn, 256, 0, stream>>>(ps1, ps2, gamma, beta, stats);
  gemm_kernel<<<NROWS / (16 * 4), 128, 0, stream>>>(yact, pw_wh, stats, pw_b, x, out);
}
